// ResidualMap_12163347382572
// MI455X (gfx1250) — matrix operand order repaired and checked by reading, not compiled, not run on hardware
//
#include <hip/hip_runtime.h>
#include <hip/hip_bf16.h>
#include <stdint.h>

// ---------------------------------------------------------------------------
// MI455X (gfx1250) implementation.
//  residual_kernel : streams 96 MB (HBM-bound stage, ~4.2us @ 23.3 TB/s),
//                    fused global min/max via uint atomics (r >= 0),
//                    global_prefetch_b8 ahead of the stream.
//  conv_v/conv_h   : separable 5-tap ROI-align pool as banded 16x32 x 32x16
//                    f16 WMMA per tile (v_wmma_f32_16x16x32_f16), tiles staged
//                    to LDS via TDM tensor_load_to_lds (inline asm, D# per
//                    CDNA5 ISA) + s_wait_tensorcnt.
//  threshold_kernel: band classification -> candidate lists.
//  nms_kernel      : 3x greedy NMS (bitonic sort + suppression), final output.
// ---------------------------------------------------------------------------

#define MAPW 1024
#define NPIX (MAPW * MAPW)
#define CAP  1024

typedef __attribute__((ext_vector_type(16))) _Float16     v16h;
typedef __attribute__((ext_vector_type(8)))  float        v8f;
typedef __attribute__((ext_vector_type(4)))  unsigned int u32x4;
typedef __attribute__((ext_vector_type(8)))  unsigned int u32x8;

// workspace layout (byte offsets); requires ~12.2 MB of workspace
#define WS_R      0u
#define WS_AV     (4u << 20)
#define WS_POOL   (8u << 20)
#define WS_CTL    (12u << 20)
#define WS_HI_IX  (WS_CTL + 256u)
#define WS_HI_SC  (WS_HI_IX + CAP * 4u)
#define WS_LO_IX  (WS_HI_SC + CAP * 4u)
#define WS_LO_SC  (WS_LO_IX + CAP * 4u)
// ctl[0]=rmin bits, ctl[1]=rmax bits, ctl[2]=pooled-max bits, ctl[3]=nHigh, ctl[4]=nLow

// ---------------------------------------------------------------------------
// TDM: 2D tile (tile_w f32 elems x tile_h rows) global -> LDS, rows packed
// contiguously.  D# per CDNA5 ISA:
//   group0: count=1 | lds_addr | global_addr[56:0] | type=2
//   group1: data_size=2(4B) | tensor_dim0/1 | tile_dim0/1 | tensor_dim0_stride
//   group2/3: zero (2D tile; tile_dim2/3/4 = 0 -> unused)
// All descriptor words are wave-uniform; readfirstlane makes that explicit so
// the "s" constraints bind cleanly to SGPR quads/octets.
// ---------------------------------------------------------------------------
__device__ __forceinline__ unsigned rfl(unsigned x) {
  return (unsigned)__builtin_amdgcn_readfirstlane((int)x);
}

__device__ __forceinline__ void tdm_load_tile(void* lds_ptr, const void* gptr,
                                              unsigned tile_w, unsigned tile_h,
                                              unsigned stride_elems) {
  const unsigned long long ga  = (unsigned long long)(size_t)gptr;
  const unsigned           lds = (unsigned)(size_t)lds_ptr;  // flat low bits = LDS offset
  const unsigned dim0 = 1u << 20, dim1 = 1u << 20;           // generous OOB dims

  u32x4 g0;
  g0[0] = rfl(1u);                                           // count=1, user descriptor
  g0[1] = rfl(lds);                                          // lds_addr
  g0[2] = rfl((unsigned)(ga & 0xFFFFFFFFull));               // global_addr lo
  g0[3] = rfl((unsigned)((ga >> 32) & 0x01FFFFFFull) | (2u << 30)); // addr hi | type=2

  u32x8 g1;
  g1[0] = rfl(2u << 16);                                     // wg_mask=0, data_size=4B
  g1[1] = rfl((dim0 & 0xFFFFu) << 16);                       // tensor_dim0[15:0]
  g1[2] = rfl(((dim0 >> 16) & 0xFFFFu) | ((dim1 & 0xFFFFu) << 16));
  g1[3] = rfl(((dim1 >> 16) & 0xFFFFu) | (tile_w << 16));    // tile_dim0
  g1[4] = rfl(tile_h & 0xFFFFu);                             // tile_dim1, tile_dim2=0
  g1[5] = rfl(stride_elems);                                 // tensor_dim0_stride lo
  g1[6] = rfl(0u);
  g1[7] = rfl(0u);

  u32x4 g2;
  g2[0] = rfl(0u); g2[1] = rfl(0u); g2[2] = rfl(0u); g2[3] = rfl(0u);
  u32x4 g3;
  g3[0] = rfl(0u); g3[1] = rfl(0u); g3[2] = rfl(0u); g3[3] = rfl(0u);

  asm volatile("tensor_load_to_lds %0, %1, %2, %3"
               :: "s"(g0), "s"(g1), "s"(g2), "s"(g3)
               : "memory");
#if defined(__has_builtin) && __has_builtin(__builtin_amdgcn_s_wait_tensorcnt)
  __builtin_amdgcn_s_wait_tensorcnt(0);
#else
  asm volatile("s_wait_tensorcnt 0x0" ::: "memory");
#endif
}

// 5 tap weights over rows/cols g-2..g+2, matching reference _interp1d exactly
// (sample offsets {-1.5,-0.5,0.5,1.5}, valid iff -1 <= c <= n, clip, lerp).
__device__ __forceinline__ void tap_weights(int g, int n, float w[5]) {
  w[0] = w[1] = w[2] = w[3] = w[4] = 0.0f;
  const float offs[4] = {-1.5f, -0.5f, 0.5f, 1.5f};
#pragma unroll
  for (int s = 0; s < 4; ++s) {
    float c = (float)g + offs[s];
    if (c < -1.0f || c > (float)n) continue;
    float cc = fminf(fmaxf(c, 0.0f), (float)(n - 1));
    int   lo = (int)floorf(cc);
    int   hi = min(lo + 1, n - 1);
    float f  = cc - (float)lo;
    int il = lo - (g - 2), ih = hi - (g - 2);
    if (il >= 0 && il < 5) w[il] += 1.0f - f;
    if (ih >= 0 && ih < 5) w[ih] += f;
  }
}

// ---------------------------------------------------------------------------
// Stage 1: residual map + global min/max.  Pure bandwidth (96 MB in, 4 MB out).
// ---------------------------------------------------------------------------
__global__ __launch_bounds__(256) void residual_kernel(
    const float* __restrict__ x1, const float* __restrict__ x2,
    const float* __restrict__ x3, float* __restrict__ r,
    unsigned int* __restrict__ ctl) {
  const int i = (blockIdx.x * 256 + threadIdx.x) * 4;
  float4 acc = make_float4(0.f, 0.f, 0.f, 0.f);
#pragma unroll
  for (int c = 0; c < 8; ++c) {
    // stream-ahead prefetch (global_prefetch_b8); pulls the next lines toward L2
    __builtin_prefetch(x1 + (size_t)c * NPIX + i + 8192, 0, 1);
    __builtin_prefetch(x2 + (size_t)c * NPIX + i + 8192, 0, 1);
    __builtin_prefetch(x3 + (size_t)c * NPIX + i + 8192, 0, 1);
    const float4 a = *reinterpret_cast<const float4*>(x1 + (size_t)c * NPIX + i);
    const float4 b = *reinterpret_cast<const float4*>(x2 + (size_t)c * NPIX + i);
    const float4 d = *reinterpret_cast<const float4*>(x3 + (size_t)c * NPIX + i);
    acc.x += fabsf(a.x - b.x) + fabsf(b.x - d.x);
    acc.y += fabsf(a.y - b.y) + fabsf(b.y - d.y);
    acc.z += fabsf(a.z - b.z) + fabsf(b.z - d.z);
    acc.w += fabsf(a.w - b.w) + fabsf(b.w - d.w);
  }
  float4 o = make_float4(0.5f * acc.x, 0.5f * acc.y, 0.5f * acc.z, 0.5f * acc.w);
  *reinterpret_cast<float4*>(r + i) = o;

  float mn = fminf(fminf(o.x, o.y), fminf(o.z, o.w));
  float mx = fmaxf(fmaxf(o.x, o.y), fmaxf(o.z, o.w));
#pragma unroll
  for (int s = 16; s > 0; s >>= 1) {   // wave32 butterfly
    mn = fminf(mn, __shfl_xor(mn, s, 32));
    mx = fmaxf(mx, __shfl_xor(mx, s, 32));
  }
  __shared__ float smn[8], smx[8];
  const int lane = threadIdx.x & 31, wave = threadIdx.x >> 5;
  if (lane == 0) { smn[wave] = mn; smx[wave] = mx; }
  __syncthreads();
  if (threadIdx.x == 0) {
    float bmn = smn[0], bmx = smx[0];
#pragma unroll
    for (int w = 1; w < 8; ++w) { bmn = fminf(bmn, smn[w]); bmx = fmaxf(bmx, smx[w]); }
    atomicMin(&ctl[0], __float_as_uint(bmn));   // r >= 0 -> uint order == float order
    atomicMax(&ctl[1], __float_as_uint(bmx));
  }
}

// ---------------------------------------------------------------------------
// Stage 2a: vertical pass.  Per wave: one 16x16 tile, D = Wv(16x32) x S(32x16)
// via v_wmma_f32_16x16x32_f16.  S rows k correspond to image rows y0+k-2.
// ---------------------------------------------------------------------------
__global__ __launch_bounds__(256) void conv_v_kernel(
    const float* __restrict__ r, const unsigned int* __restrict__ ctl,
    float* __restrict__ av) {
  __shared__ float sh[8][20 * 16];   // per-wave packed tile (<=20 rows x 16 cols)
  const int lane = threadIdx.x & 31;
  const int wave = threadIdx.x >> 5;
  const int t    = blockIdx.x * 8 + wave;
  const int y0   = (t >> 6) * 16;
  const int x0   = (t & 63) * 16;

  const float rmin = __uint_as_float(ctl[0]);
  const float rmax = __uint_as_float(ctl[1]);
  const float inv  = 1.0f / (rmax - rmin + 1e-6f);

  const int rs = max(y0 - 2, 0);
  const int re = min(y0 + 18, MAPW);
  const int o0 = rs - (y0 - 2);      // first valid k index
  const int nr = re - rs;            // rows actually loaded

  float*       shw = sh[wave];
  const float* src = r + (size_t)rs * MAPW + x0;
  tdm_load_tile(shw, src, 16u, (unsigned)nr, MAPW);

  const int m = lane & 15;           // A-row for this lane == B-column for this lane
  float w5[5];
  tap_weights(y0 + m, MAPW, w5);

  v16h afrag, bfrag;
#pragma unroll
  for (int j = 0; j < 16; ++j) {
    const int v = j >> 1, h = j & 1;
    // 16-bit A 16x32 lane layout (ISA 7.12.2)
    const int Ka = (v < 4) ? (((lane < 16) ? 0 : 8) + 2 * v + h)
                           : (((lane < 16) ? 16 : 24) + 2 * (v - 4) + h);
    const int d  = Ka - m;
    afrag[j] = (_Float16)((d >= 0 && d < 5) ? w5[d] : 0.0f);
    // 16-bit B 32x16 lane layout
    const int Kb = Ka;
    float val = 0.0f;
    if (Kb >= o0 && Kb < o0 + nr) val = (shw[(Kb - o0) * 16 + m] - rmin) * inv;
    bfrag[j] = (_Float16)val;
  }

  v8f acc = {};
  acc = __builtin_amdgcn_wmma_f32_16x16x32_f16(false, afrag, false, bfrag,
                                               (short)0, acc, false, false);

  const int n     = lane & 15;
  const int mbase = (lane < 16) ? 0 : 8;
#pragma unroll
  for (int rr = 0; rr < 8; ++rr)
    av[(size_t)(y0 + mbase + rr) * MAPW + x0 + n] = acc[rr];
}

// ---------------------------------------------------------------------------
// Stage 2b: horizontal pass.  D = A(16x32) x Wh(32x16); Wh carries the 1/16
// pooling scale.  Also block-reduces the global max of pooled.
// ---------------------------------------------------------------------------
__global__ __launch_bounds__(256) void conv_h_kernel(
    const float* __restrict__ av, float* __restrict__ pooled,
    unsigned int* __restrict__ ctl) {
  __shared__ float sh[8][16 * 20];
  __shared__ float s_wmax[8];
  const int lane = threadIdx.x & 31;
  const int wave = threadIdx.x >> 5;
  const int t    = blockIdx.x * 8 + wave;
  const int y0   = (t >> 6) * 16;
  const int x0   = (t & 63) * 16;

  const int cs = max(x0 - 2, 0);
  const int ce = min(x0 + 18, MAPW);
  const int o0 = cs - (x0 - 2);
  const int nc = ce - cs;

  float*       shw = sh[wave];
  const float* src = av + (size_t)y0 * MAPW + cs;
  tdm_load_tile(shw, src, (unsigned)nc, 16u, MAPW);

  const int m = lane & 15;
  float w5[5];
  tap_weights(x0 + m, MAPW, w5);     // column weights for output col x0+m

  v16h afrag, bfrag;
#pragma unroll
  for (int j = 0; j < 16; ++j) {
    const int v = j >> 1, h = j & 1;
    const int Ka = (v < 4) ? (((lane < 16) ? 0 : 8) + 2 * v + h)
                           : (((lane < 16) ? 16 : 24) + 2 * (v - 4) + h);
    float val = 0.0f;
    if (Ka >= o0 && Ka < o0 + nc) val = shw[m * nc + (Ka - o0)];
    afrag[j] = (_Float16)val;
    const int Kb = Ka;
    const int d  = Kb - m;
    bfrag[j] = (_Float16)((d >= 0 && d < 5) ? (w5[d] * 0.0625f) : 0.0f); // /16
  }

  v8f acc = {};
  acc = __builtin_amdgcn_wmma_f32_16x16x32_f16(false, afrag, false, bfrag,
                                               (short)0, acc, false, false);

  const int n     = lane & 15;
  const int mbase = (lane < 16) ? 0 : 8;
  float lm = 0.0f;
#pragma unroll
  for (int rr = 0; rr < 8; ++rr) {
    pooled[(size_t)(y0 + mbase + rr) * MAPW + x0 + n] = acc[rr];
    lm = fmaxf(lm, acc[rr]);
  }
#pragma unroll
  for (int s = 16; s > 0; s >>= 1) lm = fmaxf(lm, __shfl_xor(lm, s, 32));
  if (lane == 0) s_wmax[wave] = lm;
  __syncthreads();
  if (threadIdx.x == 0) {
    float bm = s_wmax[0];
#pragma unroll
    for (int w = 1; w < 8; ++w) bm = fmaxf(bm, s_wmax[w]);
    atomicMax(&ctl[2], __float_as_uint(bm));   // pooled >= 0
  }
}

// ---------------------------------------------------------------------------
// Stage 3: band thresholding -> candidate lists.
// ---------------------------------------------------------------------------
__global__ __launch_bounds__(256) void threshold_kernel(
    const float* __restrict__ pooled, unsigned int* __restrict__ ctl,
    int* __restrict__ hi_ix, float* __restrict__ hi_sc,
    int* __restrict__ lo_ix, float* __restrict__ lo_sc) {
  const int   i    = blockIdx.x * 256 + threadIdx.x;
  const float pmax = __uint_as_float(ctl[2]);
  const float lg   = pooled[i] / (pmax + 1e-6f);
  if (lg > 0.95f) {
    unsigned k = atomicAdd(&ctl[3], 1u);
    if (k < CAP) { hi_ix[k] = i; hi_sc[k] = lg; }
  } else if (lg > 0.90f && lg < 0.95f) {
    unsigned k = atomicAdd(&ctl[4], 1u);
    if (k < CAP) { lo_ix[k] = i; lo_sc[k] = lg; }
  }
}

// ---------------------------------------------------------------------------
// Stage 4: greedy NMS (single 1024-thread block = 32 wave32 waves).
// Boxes are all 4x4 at integer centers: IoU depends only on (dx,dy).
// ---------------------------------------------------------------------------
__device__ __forceinline__ float iou_pair(int ia, int ib) {
  const float dx = (float)((ia & (MAPW - 1)) - (ib & (MAPW - 1)));
  const float dy = (float)((ia >> 10) - (ib >> 10));
  const float iw = fmaxf(0.0f, 4.0f - fabsf(dx));
  const float ih = fmaxf(0.0f, 4.0f - fabsf(dy));
  const float inter = iw * ih;
  return inter / (32.0f - inter);
}

__device__ void nms_run(float* sc, int* ix, unsigned* keep, int n, float thr) {
  const int tid = threadIdx.x;
  // bitonic sort, descending by score (1024 slots, pads = -1)
  for (int k = 2; k <= CAP; k <<= 1) {
    for (int j = k >> 1; j > 0; j >>= 1) {
      __syncthreads();
      const int p = tid ^ j;
      if (p > tid) {
        const bool  up = ((tid & k) == 0);
        const float a = sc[tid], b = sc[p];
        if (up ? (a < b) : (a > b)) {
          sc[tid] = b; sc[p] = a;
          const int ti = ix[tid]; ix[tid] = ix[p]; ix[p] = ti;
        }
      }
    }
  }
  __syncthreads();
  keep[tid] = (tid < n) ? 1u : 0u;
  __syncthreads();
  for (int i = 0; i < n; ++i) {   // torchvision semantics: suppress IoU > thr
    if (keep[i] && tid > i && tid < n && keep[tid] &&
        iou_pair(ix[i], ix[tid]) > thr)
      keep[tid] = 0u;
    __syncthreads();
  }
}

__global__ __launch_bounds__(1024) void nms_kernel(
    const unsigned int* __restrict__ ctl,
    const int* __restrict__ hi_ix, const float* __restrict__ hi_sc,
    const int* __restrict__ lo_ix, const float* __restrict__ lo_sc,
    float* __restrict__ out, int out_size) {
  __shared__ float    s_sc[CAP];
  __shared__ int      s_ix[CAP];
  __shared__ unsigned s_keep[CAP];
  __shared__ int      c_ix[CAP];
  __shared__ float    c_sc[CAP];
  __shared__ int      s_m;
  const int tid = threadIdx.x;

  for (int i = tid; i < out_size; i += 1024) out[i] = 0.0f;  // deterministic fill

  // high band @ IoU 0.05
  const int nh = min((int)ctl[3], CAP);
  s_sc[tid] = (tid < nh) ? hi_sc[tid] : -1.0f;
  s_ix[tid] = (tid < nh) ? hi_ix[tid] : 0;
  __syncthreads();
  nms_run(s_sc, s_ix, s_keep, nh, 0.05f);
  if (tid == 0) {
    int m = 0;
    for (int i = 0; i < nh; ++i)
      if (s_keep[i]) { c_ix[m] = s_ix[i]; c_sc[m] = s_sc[i]; ++m; }
    s_m = m;
  }
  __syncthreads();
  const int m1 = s_m;

  // low band @ IoU 0.10
  const int nl = min((int)ctl[4], CAP);
  s_sc[tid] = (tid < nl) ? lo_sc[tid] : -1.0f;
  s_ix[tid] = (tid < nl) ? lo_ix[tid] : 0;
  __syncthreads();
  nms_run(s_sc, s_ix, s_keep, nl, 0.10f);
  if (tid == 0) {
    int m = m1;
    for (int i = 0; i < nl && m < CAP; ++i)
      if (s_keep[i]) { c_ix[m] = s_ix[i]; c_sc[m] = s_sc[i]; ++m; }
    s_m = m;
  }
  __syncthreads();
  const int mc = s_m;

  // combined (high scores > low scores, so concat order == score order) @ 0.05
  s_sc[tid] = (tid < mc) ? c_sc[tid] : -1.0f;
  s_ix[tid] = (tid < mc) ? c_ix[tid] : 0;
  __syncthreads();
  nms_run(s_sc, s_ix, s_keep, mc, 0.05f);
  if (tid == 0) {
    const int nout = out_size / 5;
    int w = 0;
    for (int i = 0; i < mc && w < nout; ++i) {
      if (!s_keep[i]) continue;
      const int   id = s_ix[i];
      const float fx = (float)(id & (MAPW - 1));
      const float fy = (float)(id >> 10);
      out[w * 5 + 0] = fx - 2.0f;
      out[w * 5 + 1] = fy - 2.0f;
      out[w * 5 + 2] = fx + 2.0f;
      out[w * 5 + 3] = fy + 2.0f;
      out[w * 5 + 4] = s_sc[i];
      ++w;
    }
  }
}

__global__ void init_kernel(unsigned int* __restrict__ ctl) {
  const int t = threadIdx.x;
  if (t < 16) ctl[t] = 0u;
  if (t == 0) ctl[0] = 0x7F800000u;   // +inf bits: identity for float-as-uint min
}

extern "C" void kernel_launch(void* const* d_in, const int* in_sizes, int n_in,
                              void* d_out, int out_size, void* d_ws, size_t ws_size,
                              hipStream_t stream) {
  (void)in_sizes; (void)n_in; (void)ws_size;
  const float* x1 = (const float*)d_in[0];
  const float* x2 = (const float*)d_in[1];
  const float* x3 = (const float*)d_in[2];
  char* ws = (char*)d_ws;
  float*        r      = (float*)(ws + WS_R);
  float*        av     = (float*)(ws + WS_AV);
  float*        pooled = (float*)(ws + WS_POOL);
  unsigned int* ctl    = (unsigned int*)(ws + WS_CTL);
  int*          hi_ix  = (int*)(ws + WS_HI_IX);
  float*        hi_sc  = (float*)(ws + WS_HI_SC);
  int*          lo_ix  = (int*)(ws + WS_LO_IX);
  float*        lo_sc  = (float*)(ws + WS_LO_SC);
  float*        out    = (float*)d_out;

  init_kernel<<<1, 32, 0, stream>>>(ctl);
  residual_kernel<<<NPIX / (256 * 4), 256, 0, stream>>>(x1, x2, x3, r, ctl);
  conv_v_kernel<<<512, 256, 0, stream>>>(r, ctl, av);        // 4096 tiles / 8 waves
  conv_h_kernel<<<512, 256, 0, stream>>>(av, pooled, ctl);
  threshold_kernel<<<NPIX / 256, 256, 0, stream>>>(pooled, ctl, hi_ix, hi_sc, lo_ix, lo_sc);
  nms_kernel<<<1, 1024, 0, stream>>>(ctl, hi_ix, hi_sc, lo_ix, lo_sc, out, out_size);
}
